// GroupStat_25864293056838
// MI455X (gfx1250) — matrix operand order repaired and checked by reading, not compiled, not run on hardware
//
#include <hip/hip_runtime.h>
#include <stdint.h>

// ---------------------------------------------------------------------------
// Radial power-spectrum (GroupStat.get_spectrum) as a one-hot WMMA GEMM.
//   spec[n, r] = (sum_p x[n,p]^2 * w[p] * [idx[p]==r]) / (count[r] + eps)
// Memory-bound: 135 MB of x @ 23.3 TB/s -> ~5.8 us floor. The scatter-reduce
// runs on the matrix pipe (v_wmma_f32_16x16x32_f16) against a precomputed,
// fragment-swizzled one-hot*weight matrix (L2-resident, 9.5 MB), with a
// per-K-step shell-tile mask exploiting block-sparsity of the radial index.
// This revision: (a) all active B-fragment loads issued before any WMMA of
// the step (one wait instead of up to 9 serialized L2 round-trips),
// (b) next step's A loads software-pipelined past the current WMMA phase,
// (c) non-temporal hints on the single-use x stream to keep the one-hot
// matrix hot in L2.
// ---------------------------------------------------------------------------

typedef __attribute__((ext_vector_type(16))) _Float16 v16h;
typedef __attribute__((ext_vector_type(8)))  float    v8f;
typedef __attribute__((ext_vector_type(4)))  float    v4f;

#define S_DIM   256
#define XDIMW   129
#define MAX_R   129
#define NIMG    1024                 // B*C = 128*8
#define NPIX    (S_DIM * XDIMW)      // 33024 pixels per image
#define KSTEPS  (NPIX / 32)          // 1032 K-steps of 32 (exact)
#define NTILES  9                    // ceil(129/16) shell tiles
#define RPAD    144                  // 9*16 padded radius dim
#define MTILES  (NIMG / 16)          // 64 image tiles
#define KBLOCKS 8
#define WPB     8                    // waves per block (256 thr, wave32)
#define NSLOTS  (KBLOCKS * WPB)      // 64 K-slots per M-tile
#define EPS_F   1e-5f

// workspace layout (bytes)
#define ACC_OFF   0
#define ACC_BYTES (NIMG * RPAD * 4)          // 589824
#define MASK_OFF  ACC_BYTES
#define BSWZ_OFF  598016                      // 256B aligned
// Bswz: [KSTEPS][NTILES][32 lanes][16 halves] f16 = 9,510,912 bytes

__global__ void zero_ws(float* __restrict__ acc, uint32_t* __restrict__ mask) {
    int i = blockIdx.x * blockDim.x + threadIdx.x;
    if (i < NIMG * RPAD) acc[i] = 0.0f;
    if (i < KSTEPS)      mask[i] = 0u;
}

// one bit per shell tile touched by each 32-pixel K-step
__global__ void build_mask(const int* __restrict__ idx, uint32_t* __restrict__ mask) {
    int p = blockIdx.x * blockDim.x + threadIdx.x;
    if (p < NPIX) {
        int t = idx[p] >> 4;                         // idx in [0,128] -> tile 0..8
        atomicOr(&mask[p >> 5], 1u << t);
    }
}

// Build the one-hot*weight matrix directly in the ISA 16-bit B-fragment
// layout for V_WMMA_*_16X16X32 (32x16, K x N):
//   lane = column n (lanes 16..31 repeat n=lane-16), half h holds
//   K = 8*(lane>=16) + (h<8 ? h : h+8). One thread emits one lane's 16 halves (32 B).
__global__ void build_bswz(const int* __restrict__ idx, const float* __restrict__ w,
                           _Float16* __restrict__ bswz) {
    int g = blockIdx.x * blockDim.x + threadIdx.x;   // ((step*9)+t)*32 + lane
    if (g >= KSTEPS * NTILES * 32) return;
    int lane = g & 31;
    int st   = g >> 5;
    int t    = st % NTILES;
    int step = st / NTILES;
    int n    = lane & 15;
    int r    = t * 16 + n;
    int kofs = (lane >= 16) ? 8 : 0;
    v16h frag;
#pragma unroll
    for (int h = 0; h < 16; ++h) {
        int p = step * 32 + kofs + (h < 8 ? h : h + 8);
        float v = 0.0f;
        if (r < MAX_R && idx[p] == r) v = w[p];
        frag[h] = (_Float16)v;
    }
    *((v16h*)bswz + g) = frag;
}

__global__ __launch_bounds__(256)
void pspec_gemm(const float* __restrict__ x, const uint32_t* __restrict__ mask,
                const _Float16* __restrict__ bswz, float* __restrict__ accum) {
    __shared__ float smem[16 * RPAD];                 // 9.2 KB partial tile
    const int tid   = threadIdx.x;
    const int lane  = tid & 31;
    const int wave  = tid >> 5;
    const int n0    = blockIdx.x * 16;                // 16 image rows
    const int slot  = blockIdx.y * WPB + wave;        // K-slot 0..63

    for (int i = tid; i < 16 * RPAD; i += 256) smem[i] = 0.0f;
    __syncthreads();

    v8f acc[NTILES];
#pragma unroll
    for (int t = 0; t < NTILES; ++t) {
        v8f z = {0.f, 0.f, 0.f, 0.f, 0.f, 0.f, 0.f, 0.f};
        acc[t] = z;
    }

    // A-fragment addressing per ISA 16-bit A layout (16x32, M x K):
    // lane<16 holds M=lane, K {0..7,16..23}; lane>=16 holds M=lane-16,
    // K {8..15,24..31}. Each lane reads 4 float4 runs; a K-step consumes
    // 16 full 128-B cachelines (100% utilization). x is single-use ->
    // non-temporal so the L2 keeps the reused one-hot matrix instead.
    const float* row = x + (size_t)(n0 + (lane & 15)) * NPIX + ((lane >= 16) ? 8 : 0);
    const v16h* bbase = (const v16h*)bswz;

    v4f q0, q1, q2, q3;
    uint32_t mnext = 0u;
    int step = slot;
    if (step < KSTEPS) {
        const float* rp = row + step * 32;
        q0 = __builtin_nontemporal_load((const v4f*)(rp + 0));
        q1 = __builtin_nontemporal_load((const v4f*)(rp + 4));
        q2 = __builtin_nontemporal_load((const v4f*)(rp + 16));
        q3 = __builtin_nontemporal_load((const v4f*)(rp + 20));
        mnext = mask[step];
    }

    while (step < KSTEPS) {
        const int nstep  = step + NSLOTS;
        const uint32_t m = (uint32_t)__builtin_amdgcn_readfirstlane((int)mnext);
        const v16h* bp   = bbase + (size_t)step * (NTILES * 32) + lane;

        // Phase 1: issue ALL active B-fragment loads for this step.
        v16h bf[NTILES];
#pragma unroll
        for (int t = 0; t < NTILES; ++t)
            if (m & (1u << t)) bf[t] = bp[t * 32];

        // Build A (square + cvt to f16) while the B loads are in flight.
        v16h a;
        a[0]  = (_Float16)(q0.x * q0.x);  a[1]  = (_Float16)(q0.y * q0.y);
        a[2]  = (_Float16)(q0.z * q0.z);  a[3]  = (_Float16)(q0.w * q0.w);
        a[4]  = (_Float16)(q1.x * q1.x);  a[5]  = (_Float16)(q1.y * q1.y);
        a[6]  = (_Float16)(q1.z * q1.z);  a[7]  = (_Float16)(q1.w * q1.w);
        a[8]  = (_Float16)(q2.x * q2.x);  a[9]  = (_Float16)(q2.y * q2.y);
        a[10] = (_Float16)(q2.z * q2.z);  a[11] = (_Float16)(q2.w * q2.w);
        a[12] = (_Float16)(q3.x * q3.x);  a[13] = (_Float16)(q3.y * q3.y);
        a[14] = (_Float16)(q3.z * q3.z);  a[15] = (_Float16)(q3.w * q3.w);

        // Software pipeline: issue next step's A stream + mask now, so its
        // HBM latency overlaps this step's WMMA phase.
        v4f p0 = q0, p1 = q1, p2 = q2, p3 = q3;
        if (nstep < KSTEPS) {
            const float* rp = row + nstep * 32;
            p0 = __builtin_nontemporal_load((const v4f*)(rp + 0));
            p1 = __builtin_nontemporal_load((const v4f*)(rp + 4));
            p2 = __builtin_nontemporal_load((const v4f*)(rp + 16));
            p3 = __builtin_nontemporal_load((const v4f*)(rp + 20));
            mnext = mask[nstep];
        }

        // Phase 2: WMMAs (one wait total for the already-issued B loads).
#pragma unroll
        for (int t = 0; t < NTILES; ++t)
            if (m & (1u << t))
                acc[t] = __builtin_amdgcn_wmma_f32_16x16x32_f16(
                    false, a, false, bf[t], (short)0, acc[t], false, false);

        q0 = p0; q1 = p1; q2 = p2; q3 = p3;
        step = nstep;
    }

    // C/D layout: lanes 0-15 VGPR j -> (M=j, N=lane); lanes 16-31 -> (M=8+j).
    const int mrow = (lane < 16) ? 0 : 8;
    const int ncol = lane & 15;
#pragma unroll
    for (int t = 0; t < NTILES; ++t) {
        const int r = t * 16 + ncol;
#pragma unroll
        for (int j = 0; j < 8; ++j)
            atomicAdd(&smem[(mrow + j) * RPAD + r], acc[t][j]);
    }
    __syncthreads();

    for (int i = tid; i < 16 * RPAD; i += 256) {
        int mm = i / RPAD, r = i % RPAD;
        atomicAdd(&accum[(size_t)(n0 + mm) * RPAD + r], smem[i]);
    }
}

__global__ void finalize(const float* __restrict__ accum,
                         const float* __restrict__ cnt, float* __restrict__ out) {
    int i = blockIdx.x * blockDim.x + threadIdx.x;
    if (i >= NIMG * MAX_R) return;
    int n = i / MAX_R, r = i % MAX_R;
    out[i] = accum[(size_t)n * RPAD + r] / (cnt[r] + EPS_F);
}

extern "C" void kernel_launch(void* const* d_in, const int* in_sizes, int n_in,
                              void* d_out, int out_size, void* d_ws, size_t ws_size,
                              hipStream_t stream) {
    const float*    x    = (const float*)d_in[0];
    const int*      idx  = (const int*)d_in[1];
    const float*    w    = (const float*)d_in[2];
    const float*    cnt  = (const float*)d_in[3];
    float*          out  = (float*)d_out;
    char*           ws   = (char*)d_ws;
    float*          acc  = (float*)(ws + ACC_OFF);
    uint32_t*       mask = (uint32_t*)(ws + MASK_OFF);
    _Float16*       bswz = (_Float16*)(ws + BSWZ_OFF);

    zero_ws<<<(NIMG * RPAD + 255) / 256, 256, 0, stream>>>(acc, mask);
    build_mask<<<(NPIX + 255) / 256, 256, 0, stream>>>(idx, mask);
    build_bswz<<<(KSTEPS * NTILES * 32 + 255) / 256, 256, 0, stream>>>(idx, w, bswz);
    pspec_gemm<<<dim3(MTILES, KBLOCKS), 256, 0, stream>>>(x, mask, bswz, acc);
    finalize<<<(NIMG * MAX_R + 255) / 256, 256, 0, stream>>>(acc, cnt, out);
}
